// CustomConv2d_8229157339513
// MI455X (gfx1250) — 8-bit matrix operand order repaired and checked by reading, compiled, not run on hardware
//
#include <hip/hip_runtime.h>

typedef int   v8i __attribute__((ext_vector_type(8)));

#define M_TOTAL 100352
#define K_TOTAL 1152
#define NB      18
#define C_IN    128
#define C_OUT   256
#define HW      56
#define OHW     3136

__device__ __forceinline__ void bfp_scales(float maxabs, float& sc, float& inv) {
    unsigned eb = (__float_as_uint(maxabs) >> 23) & 255u;
    if (eb == 0u || eb >= 255u) { sc = 1.0f; inv = 1.0f; return; }
    unsigned sb = (eb > 6u) ? (eb - 6u) : 1u;
    sc  = __uint_as_float(sb << 23);
    inv = __uint_as_float((254u - sb) << 23);
}

__device__ __forceinline__ int quant_pack4(const float* v, float inv) {
    int r = 0;
#pragma unroll
    for (int i = 0; i < 4; ++i) {
        float q = rintf(v[i] * inv);
        q = fminf(fmaxf(q, -128.0f), 127.0f);
        int qi = (int)q;
        r |= (qi & 0xff) << (8 * i);
    }
    return r;
}

__global__ void bfp_quant_weights(const float* __restrict__ w,
                                  char* __restrict__ qw, float* __restrict__ sw) {
    int t = blockIdx.x * blockDim.x + threadIdx.x;
    if (t >= NB * C_OUT) return;
    int co = t & 255;
    int kb = t >> 8;
    const float* src = w + (size_t)co * K_TOTAL + kb * 64;
    float vals[64];
    float ma = 0.0f;
#pragma unroll
    for (int i = 0; i < 64; ++i) { vals[i] = src[i]; ma = fmaxf(ma, fabsf(vals[i])); }
    float sc, inv; bfp_scales(ma, sc, inv);
    int* dq = (int*)(qw + (size_t)(kb * C_OUT + co) * 64);
#pragma unroll
    for (int g = 0; g < 16; ++g) dq[g] = quant_pack4(vals + 4 * g, inv);
    sw[kb * C_OUT + co] = sc;
}

__global__ __launch_bounds__(256, 1)
void bfp_conv_gemm(const float* __restrict__ x, const char* __restrict__ qw,
                   const float* __restrict__ sw, const float* __restrict__ bias,
                   float* __restrict__ out) {
    __shared__ int   ktab[K_TOTAL];
    __shared__ int   qaT[64 * 17];
    __shared__ int   qwT[256 * 17];
    __shared__ float sAr[64];
    __shared__ float sWr[256];
    __shared__ float pma[256];
    __shared__ float cT[8 * 16 * 64];

    const int t     = threadIdx.x;
    const int mbase = blockIdx.x * 64;

    for (int idx = t; idx < K_TOTAL; idx += 256) {
        int c  = idx / 9, r = idx - c * 9;
        int kh = r / 3,  kw = r - kh * 3;
        int off = c * OHW + (kh - 1) * HW + (kw - 1);
        ktab[idx] = ((off + 57) << 4) | (kh << 2) | kw;
    }

    const int mi   = t & 63;
    const int kseg = (t >> 6) * 16;
    const int mrow = mbase + mi;
    const int nimg = mrow / OHW;
    const int pohw = mrow - nimg * OHW;
    const int oh   = pohw / HW;
    const int ow   = pohw - oh * HW;
    const int ohm1 = oh - 1, owm1 = ow - 1;
    const float* xbase = x + (size_t)nimg * C_IN * OHW + pohw;

    const int lane = t & 31;
    const int wv   = t >> 5;
    const int mw   = wv >> 1;
    const int nh   = wv & 1;
    const int lm   = lane & 15;
    const int hi   = lane >> 4;

    float acc[8][8];
#pragma unroll
    for (int nt = 0; nt < 8; ++nt)
#pragma unroll
        for (int v = 0; v < 8; ++v) acc[nt][v] = 0.0f;

    for (int kb = 0; kb < NB; ++kb) {
        __syncthreads();

        if (kb + 1 < NB)
            __builtin_prefetch(qw + (size_t)((kb + 1) * C_OUT + t) * 64, 0, 1);

        float vals[16];
        float ma = 0.0f;
#pragma unroll
        for (int i = 0; i < 16; ++i) {
            int e  = ktab[kb * 64 + kseg + i];
            int kh = (e >> 2) & 3;
            int kw = e & 3;
            float v = 0.0f;
            if ((unsigned)(ohm1 + kh) < 56u && (unsigned)(owm1 + kw) < 56u)
                v = xbase[(e >> 4) - 57];
            vals[i] = v;
            ma = fmaxf(ma, fabsf(v));
        }
        pma[t] = ma;

        {
            const char* src = qw + (size_t)(kb * C_OUT + t) * 64;
#pragma unroll
            for (int q = 0; q < 4; ++q) {
                int4 d = *(const int4*)(src + q * 16);
                qwT[t * 17 + q * 4 + 0] = d.x;
                qwT[t * 17 + q * 4 + 1] = d.y;
                qwT[t * 17 + q * 4 + 2] = d.z;
                qwT[t * 17 + q * 4 + 3] = d.w;
            }
            sWr[t] = sw[kb * C_OUT + t];
        }
        __syncthreads();

        float m0 = fmaxf(pma[mi], pma[mi + 64]);
        float m1 = fmaxf(pma[mi + 128], pma[mi + 192]);
        ma = fmaxf(m0, m1);
        float sc, inv; bfp_scales(ma, sc, inv);
#pragma unroll
        for (int g = 0; g < 4; ++g)
            qaT[mi * 17 + (kseg >> 2) + g] = quant_pack4(vals + 4 * g, inv);
        if (kseg == 0) sAr[mi] = sc;
        __syncthreads();

        v8i afrag;
#pragma unroll
        for (int v = 0; v < 8; ++v) {
            int k0 = (v >> 1) * 16 + (v & 1) * 4 + hi * 8;
            afrag[v] = qaT[(mw * 16 + lm) * 17 + (k0 >> 2)];
        }
        float sa[8];
#pragma unroll
        for (int v = 0; v < 8; ++v) sa[v] = sAr[mw * 16 + v + hi * 8];

#pragma unroll
        for (int nt = 0; nt < 8; ++nt) {
            int col = nh * 128 + nt * 16 + lm;
            v8i bfrag;
#pragma unroll
            for (int v = 0; v < 8; ++v) {
                int k0 = (v >> 1) * 16 + (v & 1) * 4 + hi * 8;
                bfrag[v] = qwT[col * 17 + (k0 >> 2)];
            }
            float sb = sWr[col];
            v8i zero = {0, 0, 0, 0, 0, 0, 0, 0};
            v8i d = __builtin_amdgcn_wmma_i32_16x16x64_iu8(true, afrag, true, bfrag,
                                                           zero, false, false);
#pragma unroll
            for (int v = 0; v < 8; ++v)
                acc[nt][v] = fmaf((float)d[v], sa[v] * sb, acc[nt][v]);
        }
    }

    __syncthreads();
    float* cTw = cT + wv * (16 * 64);
    const int row = mbase + mw * 16 + lm;
    const int ni  = row / OHW;
    const int po  = row - ni * OHW;
#pragma unroll
    for (int g = 0; g < 2; ++g) {
#pragma unroll
        for (int n2 = 0; n2 < 4; ++n2) {
            int nt  = g * 4 + n2;
            float b = bias[nh * 128 + nt * 16 + lm];
#pragma unroll
            for (int v = 0; v < 8; ++v)
                cTw[(v + hi * 8) * 64 + n2 * 16 + lm] = acc[nt][v] + b;
        }
#pragma unroll
        for (int i = 0; i < 32; ++i) {
            int col   = nh * 128 + g * 64 + i * 2 + hi;
            float val = cTw[lm * 64 + i * 2 + hi];
            out[((size_t)ni * C_OUT + col) * OHW + po] = val;
        }
    }
}

extern "C" void kernel_launch(void* const* d_in, const int* in_sizes, int n_in,
                              void* d_out, int out_size, void* d_ws, size_t ws_size,
                              hipStream_t stream) {
    const float* x    = (const float*)d_in[0];
    const float* w    = (const float*)d_in[1];
    const float* bias = (const float*)d_in[2];
    float* out = (float*)d_out;

    char*  qw = (char*)d_ws;
    float* sw = (float*)(qw + (size_t)NB * C_OUT * 64);

    bfp_quant_weights<<<NB, 256, 0, stream>>>(w, qw, sw);
    bfp_conv_gemm<<<M_TOTAL / 64, 256, 0, stream>>>(x, qw, sw, bias, out);
}
